// NMTreeModel_41927470743633
// MI455X (gfx1250) — matrix operand order repaired and checked by reading, not compiled, not run on hardware
//
#include <hip/hip_runtime.h>

// ---------------------------------------------------------------------------
// TreeLSTM (NMTreeModel) for gfx1250: bf16 WMMA GEMMs, level-parallel scans.
// B=64, N=63 (binary heap), K=2 children, DIN=300 (padded 320), M=1024.
// GEMM stages weight tiles to LDS with GLOBAL_LOAD_ASYNC_TO_LDS_B128
// (double-buffered, ASYNCcnt-tracked), WMMAs consume B-fragments from LDS.
// ---------------------------------------------------------------------------

typedef __attribute__((ext_vector_type(16))) __bf16 v16bf;
typedef __attribute__((ext_vector_type(8)))  float  v8f;
typedef __attribute__((ext_vector_type(4)))  unsigned int u32x4;

union FragU { v16bf v; u32x4 q[2]; };

#define NB 64      // batch
#define NN 63      // nodes
#define NM 1024    // hidden
#define KX 320     // padded DIN (300 -> 320)

__device__ inline __bf16 f2bf(float f) {
  union { float f; unsigned int u; } a; a.f = f;
  unsigned int u = a.u;
  if ((u & 0x7fffffffu) > 0x7f800000u) u = 0x7fc00000u;     // NaN
  else u += 0x7fffu + ((u >> 16) & 1u);                     // RNE
  union { unsigned short s; __bf16 b; } r; r.s = (unsigned short)(u >> 16);
  return r.b;
}
__device__ inline float bf2f(__bf16 b) {
  union { unsigned short s; __bf16 b; } t; t.b = b;
  union { unsigned int u; float f; } r; r.u = ((unsigned int)t.s) << 16;
  return r.f;
}
__device__ inline float sigm(float x) { return 1.0f / (1.0f + __expf(-x)); }

// A-fragment (16x32 bf16, MxK): lane<16 -> row=lane, k in {k0..k0+7, k0+16..k0+23}
//                               lane>=16 -> row=lane-16, k in {k0+8..+15, k0+24..+31}
__device__ inline v16bf load_frag_A(const __bf16* __restrict__ p0, int lane) {
  const __bf16* p = p0 + ((lane & 16) ? 8 : 0);
  FragU f;
  f.q[0] = *(const u32x4*)(p);
  f.q[1] = *(const u32x4*)(p + 16);
  return f.v;
}

// ---------------------------------------------------------------------------
// Generic level GEMM:  C[r, c] = bias1[c]+bias2[c] + A1[r,:]·W1t[c,:] + A2[r,:]·W2t[c,:]
// Logical row r = local_node*64 + b.  Block = 64 rows (one node, all batch) x 64 cols.
// B tile (64 cols x 32 k, bf16 = 4KB) staged in LDS via async copy, double-buffered.
// ---------------------------------------------------------------------------
__device__ inline void gemm_phase(const __bf16* __restrict__ arow,
                                  const __bf16* __restrict__ Wt, int K, int colBase,
                                  __bf16 (&sB)[2][64 * 32],
                                  v8f* acc, int lane, int tid) {
  // Stage one 64x32 B tile: 256 x 16B chunks, 2 per thread, via async-to-LDS.
  auto issue = [&](int buf, int k0) {
    #pragma unroll
    for (int i = 0; i < 2; ++i) {
      const int ch   = tid + i * 128;      // 0..255
      const int row  = ch >> 2;            // 0..63  (output column)
      const int part = ch & 3;             // 16B part within the 64B row
      const __bf16* g = Wt + (long)(colBase + row) * K + k0 + part * 8;
      unsigned lds = (unsigned)(size_t)&sB[buf][row * 32 + part * 8];
      asm volatile("global_load_async_to_lds_b128 %0, %1, off"
                   :: "v"(lds), "v"((unsigned long long)(size_t)g)
                   : "memory");
    }
  };

  const int n16  = lane & 15;
  const int koff = (lane & 16) ? 8 : 0;    // B-frag: lane<16 -> k0..7,16..23, lane>=16 -> k8..15,24..31

  issue(0, 0);
  for (int k = 0; k < K; k += 32) {
    const int cur = (k >> 5) & 1;
    const bool more = (k + 32) < K;
    if (more) {
      issue(cur ^ 1, k + 32);
      asm volatile("s_wait_asynccnt 0x2" ::: "memory");  // current buffer's 2 loads done
    } else {
      asm volatile("s_wait_asynccnt 0x0" ::: "memory");
    }
    __syncthreads();                                     // all waves' chunks visible

    v16bf a = load_frag_A(arow + k, lane);
    auto loadB = [&](int rowc) -> v16bf {
      const __bf16* p = &sB[cur][rowc * 32 + koff];
      FragU f;
      f.q[0] = *(const u32x4*)(p);
      f.q[1] = *(const u32x4*)(p + 16);
      return f.v;
    };
    v16bf b0 = loadB(n16);
    v16bf b1 = loadB(n16 + 16);
    v16bf b2 = loadB(n16 + 32);
    v16bf b3 = loadB(n16 + 48);
    acc[0] = __builtin_amdgcn_wmma_f32_16x16x32_bf16(false, a, false, b0, (short)0, acc[0], false, false);
    acc[1] = __builtin_amdgcn_wmma_f32_16x16x32_bf16(false, a, false, b1, (short)0, acc[1], false, false);
    acc[2] = __builtin_amdgcn_wmma_f32_16x16x32_bf16(false, a, false, b2, (short)0, acc[2], false, false);
    acc[3] = __builtin_amdgcn_wmma_f32_16x16x32_bf16(false, a, false, b3, (short)0, acc[3], false, false);

    __syncthreads();                                     // done reading before overwrite
  }
}

__global__ void __launch_bounds__(128)
tree_gemm(float* __restrict__ Cout, int cols,
          const __bf16* __restrict__ A1, long a1_bPitch, long a1_nPitch, int a1_nodeIsLocal,
          const __bf16* __restrict__ W1t, int K1,
          const __bf16* __restrict__ A2, long a2_bPitch, long a2_nPitch,
          const __bf16* __restrict__ W2t, int K2,
          const int* __restrict__ a2_map, int a2_ms, int a2_mo,
          const float* __restrict__ bias1, const float* __restrict__ bias2,
          int node_base)
{
  __shared__ __align__(128) __bf16 sB[2][64 * 32];   // 2 x 4KB B-tile buffers

  const int tid   = threadIdx.x;
  const int lane  = tid & 31;
  const int wave  = tid >> 5;
  const int local = blockIdx.y;
  const int node  = node_base + local;
  const int colBase = blockIdx.x * 64;
  const int n16 = lane & 15;
  const int b   = wave * 16 + n16;          // batch index carried by this lane's A row

  const long a1n = a1_nodeIsLocal ? (long)local : (long)node;
  const __bf16* a1row = A1 + (long)b * a1_bPitch + a1n * a1_nPitch;

  v8f acc[4] = {};

  gemm_phase(a1row, W1t, K1, colBase, sB, acc, lane, tid);

  if (K2 > 0) {
    long nt = a2_map ? (long)a2_map[node * a2_ms + a2_mo] : (long)local;
    const __bf16* a2row = A2 + (long)b * a2_bPitch + nt * a2_nPitch;
    gemm_phase(a2row, W2t, K2, colBase, sB, acc, lane, tid);
  }

  // C/D layout: VGPR j holds M=j (lanes 0-15) / M=j+8 (lanes 16-31); N = lane&15.
  const int rowOff = (lane & 16) ? 8 : 0;
  const long rowBase = (long)local * 64 + wave * 16 + rowOff;
  #pragma unroll
  for (int t = 0; t < 4; ++t) {
    const int col = colBase + t * 16 + n16;
    float bv = 0.0f;
    if (bias1) bv += bias1[col];
    if (bias2) bv += bias2[col];
    #pragma unroll
    for (int j = 0; j < 8; ++j)
      Cout[(rowBase + j) * (long)cols + col] = acc[t][j] + bv;
  }
}

// ---------------------------------------------------------------------------
// Weight transpose + fp32->bf16 (Wt[c][kp] = W[kp][c], zero-pad kp>=K)
// ---------------------------------------------------------------------------
__global__ void conv_wt(const float* __restrict__ W, __bf16* __restrict__ Wt,
                        int K, int C, int Kp) {
  long id = (long)blockIdx.x * blockDim.x + threadIdx.x;
  long total = (long)C * Kp;
  if (id >= total) return;
  int kp = (int)(id % Kp);
  int c  = (int)(id / Kp);
  float v = (kp < K) ? W[(long)kp * C + c] : 0.0f;
  Wt[id] = f2bf(v);
}

// Inputs fp32 (B,N,300) -> bf16 padded (B,N,320)
__global__ void conv_x(const float* __restrict__ x, __bf16* __restrict__ xb) {
  long id = (long)blockIdx.x * blockDim.x + threadIdx.x;
  long total = (long)NB * NN * KX;
  if (id >= total) return;
  int kp = (int)(id % KX);
  long r = id / KX;
  xb[id] = (kp < 300) ? f2bf(x[r * 300 + kp]) : f2bf(0.0f);
}

// mask[b*63+n] = 1 if argmax(x@Wtype+btype) == 0  (lang branch)
__global__ void mask_kernel(const float* __restrict__ x, const float* __restrict__ Wtype,
                            const float* __restrict__ btype, int* __restrict__ mask) {
  int id = blockIdx.x * blockDim.x + threadIdx.x;
  if (id >= NB * NN) return;
  const float* xv = x + (long)id * 300;
  float t0 = btype[0], t1 = btype[1];
  for (int k = 0; k < 300; ++k) {
    float v = xv[k];
    t0 += v * Wtype[k * 2 + 0];
    t1 += v * Wtype[k * 2 + 1];
  }
  mask[id] = (t0 >= t1) ? 1 : 0;
}

__global__ void zero32(unsigned int* __restrict__ p, long n) {
  long id = (long)blockIdx.x * blockDim.x + threadIdx.x;
  if (id < n) p[id] = 0u;
}

// hsumb[(local*64+b)*1024+m] = bf16( h[b,ch0,m] + h[b,ch1,m] )   (sentinel 63 -> zeros)
__global__ void hsum_prep(const __bf16* __restrict__ h_up, const int* __restrict__ child_idx,
                          __bf16* __restrict__ hsumb, int node_base, int cnt) {
  long id = (long)blockIdx.x * blockDim.x + threadIdx.x;
  long total = (long)cnt * NB * NM;
  if (id >= total) return;
  int m = (int)(id & (NM - 1));
  long rb = id >> 10;
  int b = (int)(rb & 63);
  int local = (int)(rb >> 6);
  int node = node_base + local;
  int c0 = child_idx[node * 2 + 0];
  int c1 = child_idx[node * 2 + 1];
  float s = bf2f(h_up[((long)b * 64 + c0) * NM + m]) +
            bf2f(h_up[((long)b * 64 + c1) * NM + m]);
  hsumb[rb * NM + m] = f2bf(s);
}

// Up-pass epilogue: gates, cell update with 2 children, lang/vis select.
__global__ void up_epilogue(const float* __restrict__ iouL, const float* __restrict__ iouV,
                            const float* __restrict__ fL0, const float* __restrict__ fL1,
                            const float* __restrict__ fV0, const float* __restrict__ fV1,
                            float* __restrict__ c_up, __bf16* __restrict__ h_up,
                            const int* __restrict__ mask, const int* __restrict__ child_idx,
                            int node_base, int cnt) {
  long id = (long)blockIdx.x * blockDim.x + threadIdx.x;
  long total = (long)cnt * NB * NM;
  if (id >= total) return;
  int m = (int)(id & (NM - 1));
  long rb = id >> 10;
  int b = (int)(rb & 63);
  int local = (int)(rb >> 6);
  int node = node_base + local;

  long r3 = rb * 3072;
  long rf = rb * NM + m;
  int c0 = child_idx[node * 2 + 0];
  int c1 = child_idx[node * 2 + 1];
  float cc0 = c_up[((long)b * 64 + c0) * NM + m];
  float cc1 = c_up[((long)b * 64 + c1) * NM + m];

  float iL = sigm(iouL[r3 + m]);
  float oL = sigm(iouL[r3 + 1024 + m]);
  float uL = tanhf(iouL[r3 + 2048 + m]);
  float cL = iL * uL + sigm(fL0[rf]) * cc0 + sigm(fL1[rf]) * cc1;
  float hL = oL * tanhf(cL);

  float iV = sigm(iouV[r3 + m]);
  float oV = sigm(iouV[r3 + 1024 + m]);
  float uV = tanhf(iouV[r3 + 2048 + m]);
  float cV = iV * uV + sigm(fV0[rf]) * cc0 + sigm(fV1[rf]) * cc1;
  float hV = oV * tanhf(cV);

  bool lang = mask[b * NN + node] != 0;
  float c = lang ? cL : cV;
  float h = lang ? hL : hV;
  long o = ((long)b * 64 + node) * NM + m;
  c_up[o] = c;
  h_up[o] = f2bf(h);
}

// Down-pass epilogue: single parent child; also writes final output dn_h.
__global__ void down_epilogue(const float* __restrict__ iouL, const float* __restrict__ iouV,
                              const float* __restrict__ fL, const float* __restrict__ fV,
                              float* __restrict__ c_dn, __bf16* __restrict__ h_dn,
                              float* __restrict__ out,
                              const int* __restrict__ mask, const int* __restrict__ parent,
                              int node_base, int cnt) {
  long id = (long)blockIdx.x * blockDim.x + threadIdx.x;
  long total = (long)cnt * NB * NM;
  if (id >= total) return;
  int m = (int)(id & (NM - 1));
  long rb = id >> 10;
  int b = (int)(rb & 63);
  int local = (int)(rb >> 6);
  int node = node_base + local;

  long r3 = rb * 3072;
  long rf = rb * NM + m;
  int p = parent[node];
  float cp = c_dn[((long)b * 64 + p) * NM + m];

  float iL = sigm(iouL[r3 + m]);
  float oL = sigm(iouL[r3 + 1024 + m]);
  float uL = tanhf(iouL[r3 + 2048 + m]);
  float cL = iL * uL + sigm(fL[rf]) * cp;
  float hL = oL * tanhf(cL);

  float iV = sigm(iouV[r3 + m]);
  float oV = sigm(iouV[r3 + 1024 + m]);
  float uV = tanhf(iouV[r3 + 2048 + m]);
  float cV = iV * uV + sigm(fV[rf]) * cp;
  float hV = oV * tanhf(cV);

  bool lang = mask[b * NN + node] != 0;
  float c = lang ? cL : cV;
  float h = lang ? hL : hV;
  long o = ((long)b * 64 + node) * NM + m;
  c_dn[o] = c;
  h_dn[o] = f2bf(h);
  out[((long)b * NN + node) * NM + m] = h;
}

// ---------------------------------------------------------------------------
extern "C" void kernel_launch(void* const* d_in, const int* in_sizes, int n_in,
                              void* d_out, int out_size, void* d_ws, size_t ws_size,
                              hipStream_t stream) {
  (void)in_sizes; (void)n_in; (void)out_size; (void)ws_size;

  const float* inputs    = (const float*)d_in[0];
  const int*   child_idx = (const int*)d_in[1];
  const int*   parent    = (const int*)d_in[2];
  // branch param sets, insertion order: 0=up.vis 1=up.lang 2=down.vis 3=down.lang
  // within each: Wioux, bioux, Wiouh, biouh, Wfx, bfx, Wfh, bfh
  const float* bw[4][8];
  for (int s = 0; s < 4; ++s)
    for (int j = 0; j < 8; ++j)
      bw[s][j] = (const float*)d_in[3 + 8 * s + j];
  const float* Wtype = (const float*)d_in[35];
  const float* btype = (const float*)d_in[36];
  float* out = (float*)d_out;

  // ---- workspace carve ----
  char* wsp = (char*)d_ws;
  size_t off = 0;
  auto carve = [&](size_t bytes) -> void* {
    void* p = wsp + off;
    off = (off + bytes + 255) & ~(size_t)255;
    return p;
  };
  __bf16 *WiouxT[4], *WiouhT[4], *WfxT[4], *WfhT[4];
  for (int s = 0; s < 4; ++s) {
    WiouxT[s] = (__bf16*)carve((size_t)3072 * KX * 2);
    WiouhT[s] = (__bf16*)carve((size_t)3072 * NM * 2);
    WfxT[s]   = (__bf16*)carve((size_t)NM * KX * 2);
    WfhT[s]   = (__bf16*)carve((size_t)NM * NM * 2);
  }
  __bf16* xb   = (__bf16*)carve((size_t)NB * NN * KX * 2);
  int*    mask = (int*)   carve((size_t)NB * NN * 4);
  float*  c_up = (float*) carve((size_t)NB * 64 * NM * 4);
  __bf16* h_up = (__bf16*)carve((size_t)NB * 64 * NM * 2);
  float*  c_dn = (float*) carve((size_t)NB * 64 * NM * 4);
  __bf16* h_dn = (__bf16*)carve((size_t)NB * 64 * NM * 2);
  __bf16* hsumb= (__bf16*)carve((size_t)32 * NB * NM * 2);
  float*  iouL = (float*) carve((size_t)32 * NB * 3072 * 4);
  float*  iouV = (float*) carve((size_t)32 * NB * 3072 * 4);
  float*  fb[4];
  for (int i = 0; i < 4; ++i) fb[i] = (float*)carve((size_t)32 * NB * NM * 4);

  // ---- one-time prep: weights bf16+transpose, inputs bf16, mask, zero states ----
  auto launchT = [&](const float* W, __bf16* Wt, int K, int C, int Kp) {
    long total = (long)C * Kp;
    conv_wt<<<(unsigned)((total + 255) / 256), 256, 0, stream>>>(W, Wt, K, C, Kp);
  };
  for (int s = 0; s < 4; ++s) {
    launchT(bw[s][0], WiouxT[s], 300,  3072, KX);
    launchT(bw[s][2], WiouhT[s], 1024, 3072, NM);
    launchT(bw[s][4], WfxT[s],   300,  NM,   KX);
    launchT(bw[s][6], WfhT[s],   1024, NM,   NM);
  }
  {
    long tot = (long)NB * NN * KX;
    conv_x<<<(unsigned)((tot + 255) / 256), 256, 0, stream>>>(inputs, xb);
    mask_kernel<<<(NB * NN + 255) / 256, 256, 0, stream>>>(inputs, Wtype, btype, mask);
    long nw = (long)NB * 64 * NM;                       // words per fp32 state array
    zero32<<<(unsigned)((nw + 255) / 256), 256, 0, stream>>>((unsigned int*)c_up, nw);
    zero32<<<(unsigned)((nw + 255) / 256), 256, 0, stream>>>((unsigned int*)c_dn, nw);
    zero32<<<(unsigned)((nw / 2 + 255) / 256), 256, 0, stream>>>((unsigned int*)h_up, nw / 2);
    zero32<<<(unsigned)((nw / 2 + 255) / 256), 256, 0, stream>>>((unsigned int*)h_dn, nw / 2);
  }

  auto gemm = [&](float* C, int cols,
                  const __bf16* W1, const __bf16* A2, const __bf16* W2,
                  const int* map, int ms, int mo,
                  const float* b1, const float* b2, int base, int cnt) {
    dim3 g((unsigned)(cols / 64), (unsigned)cnt);
    tree_gemm<<<g, 128, 0, stream>>>(
        C, cols,
        xb, (long)NN * KX, (long)KX, /*a1_nodeIsLocal=*/0, W1, KX,
        A2,
        /*a2_bPitch=*/(A2 == hsumb) ? (long)NM : (long)64 * NM,
        /*a2_nPitch=*/(A2 == hsumb) ? (long)64 * NM : (long)NM,
        W2, NM, map, ms, mo, b1, b2, base);
  };

  // ---- up pass: levels 5..0 (leaves -> root) ----
  for (int d = 5; d >= 0; --d) {
    int base = (1 << d) - 1;
    int cnt  = (NN - base < (1 << d)) ? (NN - base) : (1 << d);
    long tot = (long)cnt * NB * NM;
    hsum_prep<<<(unsigned)(tot / 256), 256, 0, stream>>>(h_up, child_idx, hsumb, base, cnt);
    // iou = x@Wioux + hsum@Wiouh + bioux + biouh   (lang = set 1, vis = set 0)
    gemm(iouL, 3072, WiouxT[1], hsumb, WiouhT[1], nullptr, 0, 0, bw[1][1], bw[1][3], base, cnt);
    gemm(iouV, 3072, WiouxT[0], hsumb, WiouhT[0], nullptr, 0, 0, bw[0][1], bw[0][3], base, cnt);
    // f_k = x@Wfx + h[child_k]@Wfh + bfx + bfh
    for (int k = 0; k < 2; ++k) {
      gemm(fb[k],     NM, WfxT[1], h_up, WfhT[1], child_idx, 2, k, bw[1][5], bw[1][7], base, cnt);
      gemm(fb[2 + k], NM, WfxT[0], h_up, WfhT[0], child_idx, 2, k, bw[0][5], bw[0][7], base, cnt);
    }
    up_epilogue<<<(unsigned)(tot / 256), 256, 0, stream>>>(
        iouL, iouV, fb[0], fb[1], fb[2], fb[3], c_up, h_up, mask, child_idx, base, cnt);
  }

  // ---- down pass: levels 0..5 (root -> leaves); lang = set 3, vis = set 2 ----
  for (int d = 0; d <= 5; ++d) {
    int base = (1 << d) - 1;
    int cnt  = (NN - base < (1 << d)) ? (NN - base) : (1 << d);
    long tot = (long)cnt * NB * NM;
    gemm(iouL, 3072, WiouxT[3], h_dn, WiouhT[3], parent, 1, 0, bw[3][1], bw[3][3], base, cnt);
    gemm(iouV, 3072, WiouxT[2], h_dn, WiouhT[2], parent, 1, 0, bw[2][1], bw[2][3], base, cnt);
    gemm(fb[0],  NM, WfxT[3],   h_dn, WfhT[3],   parent, 1, 0, bw[3][5], bw[3][7], base, cnt);
    gemm(fb[1],  NM, WfxT[2],   h_dn, WfhT[2],   parent, 1, 0, bw[2][5], bw[2][7], base, cnt);
    down_epilogue<<<(unsigned)(tot / 256), 256, 0, stream>>>(
        iouL, iouV, fb[0], fb[1], c_dn, h_dn, out, mask, parent, base, cnt);
  }
}
